// TrajLSTM_4982162063812
// MI455X (gfx1250) — compile-verified, matrix operand order checked by reading, not run on hardware
//
#include <hip/hip_runtime.h>

typedef __attribute__((ext_vector_type(16))) _Float16 v16h;
typedef __attribute__((ext_vector_type(8)))  _Float16 v8h;
typedef __attribute__((ext_vector_type(8)))  float    v8f;

constexpr int BATCH   = 8192;
constexpr int TSEQ    = 32;
constexpr int H       = 256;
constexpr int BB      = 32;        // batch rows per block
constexpr int HSTR    = H + 8;     // padded LDS row stride (halves) -> 528B, conflict-free
constexpr int KTILES  = H / 32;    // 8 k-steps of 32 for 16x16x32 WMMA
constexpr int THREADS = 512;       // 16 waves

// ---------------- weight packing: fp32 [1024][256] -> f16 WMMA-B tile order -------------
// tile (gnt in 0..63, kt in 0..7): lane l holds column gnt*16+(l&15);
// halves q: lanes<16 -> K = kt*32 + (q<8 ? q : q+8); lanes>=16 -> +8.
__global__ void pack_w_kernel(const float* __restrict__ W, _Float16* __restrict__ pk) {
  const int idx  = blockIdx.x * blockDim.x + threadIdx.x;   // 0 .. 262143
  const int q    = idx & 15;
  const int lane = (idx >> 4) & 31;
  const int kt   = (idx >> 9) & 7;
  const int nt   = idx >> 12;
  if (nt >= 64) return;
  const int col  = nt * 16 + (lane & 15);
  const int koff = (q < 8 ? q : q + 8) + ((lane >> 4) << 3);
  const int k    = kt * 32 + koff;
  pk[idx] = (_Float16)W[col * H + k];
}

__global__ void bias_sum_kernel(const float* __restrict__ a, const float* __restrict__ b,
                                float* __restrict__ o) {
  const int i = blockIdx.x * blockDim.x + threadIdx.x;
  if (i < 1024) o[i] = a[i] + b[i];
}

// ---------------- device helpers ----------------
__device__ __forceinline__ float sigm(float x) { return 1.0f / (1.0f + __expf(-x)); }
__device__ __forceinline__ float tanh_fast(float x) { return 2.0f * sigm(2.0f * x) - 1.0f; }

__device__ __forceinline__ v16h load_a(const _Float16* shh, int mrow, int kt, int lane) {
  // 16-bit A 16x32 lane layout: lane m holds row m%16; lanes<16 K {0..7,16..23}, lanes>=16 +8
  const int r  = (mrow << 4) | (lane & 15);
  const int k0 = kt * 32 + ((lane >> 4) << 3);
  const _Float16* p = shh + r * HSTR + k0;
  union { v16h v; v8h h[2]; } u;
  u.h[0] = *(const v8h*)(p);
  u.h[1] = *(const v8h*)(p + 16);
  return u.v;
}

__device__ __forceinline__ v16h load_b(const _Float16* __restrict__ pk, int gnt, int kt, int lane) {
  return *(const v16h*)(pk + (((gnt * KTILES + kt) * 32 + lane) << 4));
}

// first WMMA of a batch: no operand-reuse hint
__device__ __forceinline__ v8f wmma_f(const v16h a, const v16h b, v8f c) {
  return __builtin_amdgcn_wmma_f32_16x16x32_f16(false, a, false, b, (short)0, c, false, false);
}
// subsequent WMMAs: same opcode, same A operand as previous -> legal reuse_a hint (ISA 7.12 RA)
__device__ __forceinline__ v8f wmma_ra(const v16h a, const v16h b, v8f c) {
  return __builtin_amdgcn_wmma_f32_16x16x32_f16(false, a, false, b, (short)0, c, true, false);
}

// batched 8-tile GEMM step: load all B tiles first (one big load clause, 16 b128 in
// flight), then drain with 8 WMMAs sharing the A operand.
__device__ __forceinline__ void gemm_batch8(const v16h a, const _Float16* __restrict__ pk,
                                            int ng, int kt, int lane, v8f acc[4][2]) {
  v16h bt[8];
  #pragma unroll
  for (int i = 0; i < 8; ++i)
    bt[i] = load_b(pk, (i >> 1) * 16 + ng * 2 + (i & 1), kt, lane);
  acc[0][0] = wmma_f(a, bt[0], acc[0][0]);
  #pragma unroll
  for (int i = 1; i < 8; ++i)
    acc[i >> 1][i & 1] = wmma_ra(a, bt[i], acc[i >> 1][i & 1]);
}

// ---------------- fused seq2seq LSTM ----------------
__global__ __launch_bounds__(THREADS)
void traj_lstm_fused(const float* __restrict__ past, const float* __restrict__ fut,
                     const _Float16* __restrict__ pkEWhh0, const _Float16* __restrict__ pkEWih1,
                     const _Float16* __restrict__ pkEWhh1,
                     const _Float16* __restrict__ pkDWhh0, const _Float16* __restrict__ pkDWih1,
                     const _Float16* __restrict__ pkDWhh1,
                     const float* __restrict__ bE0, const float* __restrict__ bE1,
                     const float* __restrict__ bD0, const float* __restrict__ bD1,
                     const float* __restrict__ wihE0, const float* __restrict__ wihD0,
                     const float* __restrict__ outW, const float* __restrict__ outB,
                     float* __restrict__ out) {
  __shared__ __align__(16) _Float16 sh0[BB * HSTR];   // layer-0 h (f16)
  __shared__ __align__(16) _Float16 sh1[BB * HSTR];   // layer-1 h (f16)
  __shared__ float sx[BB][2];

  const int tid  = threadIdx.x;
  const int lane = tid & 31;
  const int w    = tid >> 5;
  const int mrow = w >> 3;        // 0..1  : 16-row batch tile
  const int ng   = w & 7;         // 0..7  : 32-col group per gate
  const int lcol = lane & 15;
  const int lh   = lane >> 4;
  const int b0r  = blockIdx.x * BB;

  for (int i = tid; i < BB * HSTR; i += THREADS) { sh0[i] = (_Float16)0.f; sh1[i] = (_Float16)0.f; }

  float c0st[2][8], c1st[2][8];
  #pragma unroll
  for (int n = 0; n < 2; ++n)
    #pragma unroll
    for (int v = 0; v < 8; ++v) { c0st[n][v] = 0.f; c1st[n][v] = 0.f; }

  __syncthreads();

  for (int ph = 0; ph < 2; ++ph) {
    const _Float16* pWhh0 = ph ? pkDWhh0 : pkEWhh0;
    const _Float16* pWih1 = ph ? pkDWih1 : pkEWih1;
    const _Float16* pWhh1 = ph ? pkDWhh1 : pkEWhh1;
    const float* bias0 = ph ? bD0 : bE0;
    const float* bias1 = ph ? bD1 : bE1;
    const float* wih0  = ph ? wihD0 : wihE0;

    float bi0[4][2], bi1[4][2], wxa[4][2], wxb[4][2];
    #pragma unroll
    for (int g = 0; g < 4; ++g)
      #pragma unroll
      for (int n = 0; n < 2; ++n) {
        const int col = g * 256 + ng * 32 + n * 16 + lcol;
        bi0[g][n] = bias0[col];
        bi1[g][n] = bias1[col];
        wxa[g][n] = wih0[col * 2 + 0];
        wxb[g][n] = wih0[col * 2 + 1];
      }

    for (int t = 0; t < TSEQ; ++t) {
      if (tid < BB) {                                   // stage x_t
        const int b = b0r + tid;
        float x0, x1;
        if (ph == 0)      { x0 = past[(b * TSEQ + t) * 2];        x1 = past[(b * TSEQ + t) * 2 + 1]; }
        else if (t == 0)  { x0 = past[(b * TSEQ + TSEQ - 1) * 2]; x1 = past[(b * TSEQ + TSEQ - 1) * 2 + 1]; }
        else              { x0 = fut[(b * TSEQ + t - 1) * 2];     x1 = fut[(b * TSEQ + t - 1) * 2 + 1]; }
        sx[tid][0] = x0; sx[tid][1] = x1;
      }
      __syncthreads();

      // ---- layer 0: gates = x@Wih0^T + bias + h0@Whh0^T ----
      v8f zero = {};
      v8f acc[4][2];
      #pragma unroll
      for (int g = 0; g < 4; ++g) { acc[g][0] = zero; acc[g][1] = zero; }

      #pragma unroll 2
      for (int kt = 0; kt < KTILES; ++kt) {
        const v16h a = load_a(sh0, mrow, kt, lane);
        gemm_batch8(a, pWhh0, ng, kt, lane, acc);
      }
      __syncthreads();                                  // all waves done reading h0(t-1)

      #pragma unroll
      for (int n = 0; n < 2; ++n)
        #pragma unroll
        for (int v = 0; v < 8; ++v) {
          const int r  = v + 8 * lh;
          const float xa = sx[mrow * 16 + r][0];
          const float xb = sx[mrow * 16 + r][1];
          const float gi = acc[0][n][v] + bi0[0][n] + xa * wxa[0][n] + xb * wxb[0][n];
          const float gf = acc[1][n][v] + bi0[1][n] + xa * wxa[1][n] + xb * wxb[1][n];
          const float gg = acc[2][n][v] + bi0[2][n] + xa * wxa[2][n] + xb * wxb[2][n];
          const float go = acc[3][n][v] + bi0[3][n] + xa * wxa[3][n] + xb * wxb[3][n];
          const float cn = sigm(gf) * c0st[n][v] + sigm(gi) * tanh_fast(gg);
          c0st[n][v] = cn;
          const float hn = sigm(go) * tanh_fast(cn);
          sh0[(mrow * 16 + r) * HSTR + ng * 32 + n * 16 + lcol] = (_Float16)hn;
        }
      __syncthreads();                                  // sh0 = h0(t)

      // ---- layer 1: gates = h0(t)@Wih1^T + bias + h1(t-1)@Whh1^T ----
      #pragma unroll
      for (int g = 0; g < 4; ++g) { acc[g][0] = zero; acc[g][1] = zero; }

      #pragma unroll 2
      for (int kt = 0; kt < KTILES; ++kt) {
        const v16h a0 = load_a(sh0, mrow, kt, lane);    // h0(t) contribution
        gemm_batch8(a0, pWih1, ng, kt, lane, acc);
        const v16h a1 = load_a(sh1, mrow, kt, lane);    // h1(t-1) recurrence
        gemm_batch8(a1, pWhh1, ng, kt, lane, acc);
      }
      __syncthreads();                                  // done reading h1(t-1)

      #pragma unroll
      for (int n = 0; n < 2; ++n)
        #pragma unroll
        for (int v = 0; v < 8; ++v) {
          const int r  = v + 8 * lh;
          const float gi = acc[0][n][v] + bi1[0][n];
          const float gf = acc[1][n][v] + bi1[1][n];
          const float gg = acc[2][n][v] + bi1[2][n];
          const float go = acc[3][n][v] + bi1[3][n];
          const float cn = sigm(gf) * c1st[n][v] + sigm(gi) * tanh_fast(gg);
          c1st[n][v] = cn;
          const float hn = sigm(go) * tanh_fast(cn);
          sh1[(mrow * 16 + r) * HSTR + ng * 32 + n * 16 + lcol] = (_Float16)hn;
        }
      __syncthreads();                                  // sh1 = h1(t)

      // ---- decoder output projection: out[b][t][j] = h1 . outW[j] + outB[j] ----
      if (ph == 1 && tid < 2 * BB) {
        const int row = tid >> 1, j = tid & 1;
        float s = outB[j];
        const _Float16* hr = &sh1[row * HSTR];
        #pragma unroll 8
        for (int k = 0; k < H; ++k) s += (float)hr[k] * outW[j * H + k];
        out[((b0r + row) * TSEQ + t) * 2 + j] = s;
      }
    }
  }
}

// ---------------- host launch ----------------
extern "C" void kernel_launch(void* const* d_in, const int* in_sizes, int n_in,
                              void* d_out, int out_size, void* d_ws, size_t ws_size,
                              hipStream_t stream) {
  const float* past  = (const float*)d_in[0];
  const float* fut   = (const float*)d_in[1];
  const float* eWih0 = (const float*)d_in[2];
  const float* eWhh0 = (const float*)d_in[3];
  const float* eBih0 = (const float*)d_in[4];
  const float* eBhh0 = (const float*)d_in[5];
  const float* eWih1 = (const float*)d_in[6];
  const float* eWhh1 = (const float*)d_in[7];
  const float* eBih1 = (const float*)d_in[8];
  const float* eBhh1 = (const float*)d_in[9];
  const float* dWih0 = (const float*)d_in[10];
  const float* dWhh0 = (const float*)d_in[11];
  const float* dBih0 = (const float*)d_in[12];
  const float* dBhh0 = (const float*)d_in[13];
  const float* dWih1 = (const float*)d_in[14];
  const float* dWhh1 = (const float*)d_in[15];
  const float* dBih1 = (const float*)d_in[16];
  const float* dBhh1 = (const float*)d_in[17];
  const float* outW  = (const float*)d_in[18];
  const float* outB  = (const float*)d_in[19];

  char* ws = (char*)d_ws;
  const size_t PKB = (size_t)64 * 8 * 32 * 16 * sizeof(_Float16);  // 512 KB per matrix
  _Float16* pkEWhh0 = (_Float16*)(ws + 0 * PKB);
  _Float16* pkEWih1 = (_Float16*)(ws + 1 * PKB);
  _Float16* pkEWhh1 = (_Float16*)(ws + 2 * PKB);
  _Float16* pkDWhh0 = (_Float16*)(ws + 3 * PKB);
  _Float16* pkDWih1 = (_Float16*)(ws + 4 * PKB);
  _Float16* pkDWhh1 = (_Float16*)(ws + 5 * PKB);
  float* bE0 = (float*)(ws + 6 * PKB);
  float* bE1 = bE0 + 1024;
  float* bD0 = bE1 + 1024;
  float* bD1 = bD0 + 1024;

  const dim3 pg(1024), pb(256);
  pack_w_kernel<<<pg, pb, 0, stream>>>(eWhh0, pkEWhh0);
  pack_w_kernel<<<pg, pb, 0, stream>>>(eWih1, pkEWih1);
  pack_w_kernel<<<pg, pb, 0, stream>>>(eWhh1, pkEWhh1);
  pack_w_kernel<<<pg, pb, 0, stream>>>(dWhh0, pkDWhh0);
  pack_w_kernel<<<pg, pb, 0, stream>>>(dWih1, pkDWih1);
  pack_w_kernel<<<pg, pb, 0, stream>>>(dWhh1, pkDWhh1);
  bias_sum_kernel<<<4, 256, 0, stream>>>(eBih0, eBhh0, bE0);
  bias_sum_kernel<<<4, 256, 0, stream>>>(eBih1, eBhh1, bE1);
  bias_sum_kernel<<<4, 256, 0, stream>>>(dBih0, dBhh0, bD0);
  bias_sum_kernel<<<4, 256, 0, stream>>>(dBih1, dBhh1, bD1);

  traj_lstm_fused<<<BATCH / BB, THREADS, 0, stream>>>(
      past, fut, pkEWhh0, pkEWih1, pkEWhh1, pkDWhh0, pkDWih1, pkDWhh1,
      bE0, bE1, bD0, bD1, eWih0, dWih0, outW, outB, (float*)d_out);
}
